// SpeechEncDec_86328842649792
// MI455X (gfx1250) — matrix operand order repaired and checked by reading, not compiled, not run on hardware
//
#include <hip/hip_runtime.h>
#include <hip/hip_bf16.h>
#include <math.h>

// SpeechEncDec for MI455X (gfx1250, wave32).
// Front-end is rank-1-collapsed; decoder input GEMMs use v_wmma_f32_16x16x32_f16
// (f32 accumulate); all recurrent chains run fp32 in a single 768-thread block
// (24 wave32) with LDS-resident state and transposed w_hh for coalesced matvecs.
// Workspace requirement: ~3.1 MB.

typedef __attribute__((ext_vector_type(16))) _Float16 v16h;
typedef __attribute__((ext_vector_type(8)))  _Float16 v8h;
typedef __attribute__((ext_vector_type(8)))  float    v8f;

#define HD 256   // hidden
#define GD 768   // 3*HD gates
#define TT 256   // sequence length

__device__ __forceinline__ float sigm(float x) { return 1.0f / (1.0f + __expf(-x)); }

// ---------------------------------------------------------------------------
// K0: rank-1 collapse of the conv front-end + encoder input projection.
//   u = w2@w1 ; v = w2@b1 + b2 ; P = enc_w_ih@u ; Q = enc_w_ih@v + enc_b_ih
// One block, 768 threads.
// ---------------------------------------------------------------------------
__global__ void front_kernel(const float* __restrict__ w1, const float* __restrict__ b1,
                             const float* __restrict__ w2, const float* __restrict__ b2,
                             const float* __restrict__ ewih, const float* __restrict__ ebih,
                             float* __restrict__ P, float* __restrict__ Q) {
  __shared__ float u[HD], v[HD];
  int tid = threadIdx.x;
  if (tid < HD) {
    float su = 0.f, sv = 0.f;
    for (int k = 0; k < 128; ++k) {
      float w = w2[tid * 128 + k];
      su += w * w1[k];
      sv += w * b1[k];
    }
    u[tid] = su;
    v[tid] = sv + b2[tid];
  }
  __syncthreads();
  float sp = 0.f, sq = 0.f;
  const float* row = ewih + tid * HD;
  for (int d = 0; d < HD; ++d) {
    float w = row[d];
    sp += w * u[d];
    sq += w * v[d];
  }
  P[tid] = sp;
  Q[tid] = sq + ebih[tid];
}

// ---------------------------------------------------------------------------
// K1: one-time weight prep.
//   WhhT_{enc,dec}[k*768+g] = w_hh[g*256+k]  (coalesced recurrent matvec)
//   Wih_h = f16(dec_w_ih)                     (WMMA B operand)
// ---------------------------------------------------------------------------
__global__ void prep_kernel(const float* __restrict__ ewhh, const float* __restrict__ dwhh,
                            const float* __restrict__ dwih,
                            float* __restrict__ WhhT_e, float* __restrict__ WhhT_d,
                            _Float16* __restrict__ Wih_h) {
  int idx = blockIdx.x * blockDim.x + threadIdx.x;
  if (idx >= GD * HD) return;
  int g = idx / HD, k = idx % HD;
  WhhT_e[k * GD + g] = ewhh[idx];
  WhhT_d[k * GD + g] = dwhh[idx];
  Wih_h[idx] = (_Float16)dwih[idx];
}

// ---------------------------------------------------------------------------
// K2: encoder GRU, 256 sequential steps, single block (768 threads = 24 wave32).
// Input projection is rank-1: xp[t,g] = X[2t]*P[g] + Q[g].
// Writes Xc_h = f16(relu(h_t)) (decoder's first GEMM A-matrix) and h_enc.
// ---------------------------------------------------------------------------
__global__ void encoder_kernel(const float* __restrict__ X,
                               const float* __restrict__ Pg, const float* __restrict__ Qg,
                               const float* __restrict__ WhhT, const float* __restrict__ bhh,
                               _Float16* __restrict__ Xc_h, float* __restrict__ Hbuf) {
  __shared__ float h[HD];
  __shared__ float gh[GD];
  __shared__ float Pl[GD], Ql[GD];
  int tid = threadIdx.x;
  Pl[tid] = Pg[tid];
  Ql[tid] = Qg[tid];
  if (tid < HD) h[tid] = 0.f;
  float bg = bhh[tid];
  for (int t = 0; t < TT; ++t) {
    __syncthreads();
    float acc = bg;
    for (int k = 0; k < HD; ++k) acc += WhhT[k * GD + tid] * h[k];  // coalesced over tid
    gh[tid] = acc;
    __syncthreads();
    if (tid < HD) {
      float xv = X[2 * t];
      float r = sigm(xv * Pl[tid] + Ql[tid] + gh[tid]);
      float z = sigm(xv * Pl[HD + tid] + Ql[HD + tid] + gh[HD + tid]);
      float n = tanhf(xv * Pl[2 * HD + tid] + Ql[2 * HD + tid] + r * gh[2 * HD + tid]);
      float hn = (1.f - z) * n + z * h[tid];
      h[tid] = hn;
      float xc = hn > 0.f ? hn : 0.f;           // relu(enc_out)
      Xc_h[t * HD + tid] = (_Float16)xc;
    }
  }
  __syncthreads();
  if (tid < HD) Hbuf[tid] = h[tid];
}

// ---------------------------------------------------------------------------
// K3: decoder input-projection GEMM via WMMA f16 (f32 accumulate).
//   XP[256,768] = Xc[256,256] @ dec_w_ih^T + dec_b_ih
// One wave per 16x16 output tile; 8 waves/block; 96 blocks; K-loop of 8
// v_wmma_f32_16x16x32_f16 per tile. EXEC is all-ones (no divergence).
// Fragment layouts used here (A and B share contiguous k order):
//   A 16x32 f16 : lane L -> m=L%16 ; K = k0 + 16*half + e
//   B 32x16 f16 : lane L -> n=L%16 ; K = k0 + 16*half + e   (B[d,g] = Wih[g,d])
//   D 16x16 f32 : VGPR r -> m = r + 8*half ; n = L%16
// ---------------------------------------------------------------------------
__global__ void dec_gemm_kernel(const _Float16* __restrict__ A,     // Xc_h [256,256]
                                const _Float16* __restrict__ Bh,    // Wih_h [768,256]
                                const float* __restrict__ bias,     // dec_b_ih [768]
                                float* __restrict__ XP) {           // [256,768]
  int wave = threadIdx.x >> 5;
  int lane = threadIdx.x & 31;
  int tile = blockIdx.x * 8 + wave;      // 0..767
  int mb = tile / 48;                    // 16 row tiles
  int nb = tile % 48;                    // 48 col tiles
  int half = lane >> 4;
  int ln = lane & 15;
  int g = nb * 16 + ln;                  // gate column

  v8f c;
  float bv = bias[g];
#pragma unroll
  for (int r = 0; r < 8; ++r) c[r] = bv;

#pragma unroll
  for (int kb = 0; kb < 8; ++kb) {
    int k0 = kb * 32;
    const _Float16* arow = A + ln * HD + k0 + 16 * half;
    v8h alo = *(const v8h*)(arow);
    v8h ahi = *(const v8h*)(arow + 8);
    v16h a;
#pragma unroll
    for (int e = 0; e < 8; ++e) { a[e] = alo[e]; a[8 + e] = ahi[e]; }
    v16h b = *(const v16h*)(Bh + g * HD + k0 + 16 * half);
    c = __builtin_amdgcn_wmma_f32_16x16x32_f16(false, a, false, b, (short)0, c, false, false);
  }

  float* out = XP + (mb * 16 + 8 * half) * GD + g;
#pragma unroll
  for (int r = 0; r < 8; ++r) out[r * GD] = c[r];
}

// ---------------------------------------------------------------------------
// K4: one decoder outer step: 256-step inner GRU (sequential, LDS state),
// then attention + fc + softmax, writing preds[s]. Single block, 768 threads.
// ---------------------------------------------------------------------------
__global__ void dec_step_kernel(const float* __restrict__ XP, const float* __restrict__ WhhT,
                                const float* __restrict__ bhh, float* __restrict__ DecOut,
                                _Float16* __restrict__ Xc_h, float* __restrict__ Hbuf,
                                const float* __restrict__ fcw, const float* __restrict__ fcb,
                                float* __restrict__ preds, int s) {
  __shared__ float h[HD];
  __shared__ float gh[GD];
  __shared__ float att[TT];
  __shared__ float ct[HD];
  __shared__ float red[2];
  __shared__ float logits[26];
  int tid = threadIdx.x;
  if (tid < HD) h[tid] = Hbuf[tid];
  float bg = bhh[tid];

  for (int t = 0; t < TT; ++t) {
    __syncthreads();
    float acc = bg;
    for (int k = 0; k < HD; ++k) acc += WhhT[k * GD + tid] * h[k];
    gh[tid] = acc;
    __syncthreads();
    if (tid < HD) {
      const float* xp = XP + t * GD;
      float r = sigm(xp[tid] + gh[tid]);
      float z = sigm(xp[HD + tid] + gh[HD + tid]);
      float n = tanhf(xp[2 * HD + tid] + r * gh[2 * HD + tid]);
      float hn = (1.f - z) * n + z * h[tid];
      h[tid] = hn;
      DecOut[t * HD + tid] = hn;
      Xc_h[t * HD + tid] = (_Float16)hn;   // next outer step's GEMM A-matrix
    }
  }
  __threadfence_block();
  __syncthreads();

  // attention: a = softmax(DecOut @ h_new)
  if (tid < TT) {
    const float* row = DecOut + tid * HD;
    float sc = 0.f;
    for (int k = 0; k < HD; ++k) sc += row[k] * h[k];
    att[tid] = sc;
  }
  __syncthreads();
  if (tid == 0) {
    float mx = att[0];
    for (int i = 1; i < TT; ++i) mx = att[i] > mx ? att[i] : mx;
    red[0] = mx;
  }
  __syncthreads();
  if (tid < TT) att[tid] = __expf(att[tid] - red[0]);
  __syncthreads();
  if (tid == 0) {
    float sm = 0.f;
    for (int i = 0; i < TT; ++i) sm += att[i];
    red[1] = 1.0f / sm;
  }
  __syncthreads();
  if (tid < TT) att[tid] *= red[1];
  __syncthreads();

  // c_t = a[0]*dec_out[0] + sum_{j=0..254} a[j]*dec_out[j+1]
  if (tid < HD) {
    float c = att[0] * DecOut[tid];
    for (int j = 0; j < TT - 1; ++j) c += att[j] * DecOut[(j + 1) * HD + tid];
    ct[tid] = c;
  }
  __syncthreads();

  // o = softmax(fc_w @ [h_new; c_t] + fc_b)
  if (tid < 26) {
    float acc = fcb[tid];
    const float* wr = fcw + tid * 512;
    for (int i = 0; i < HD; ++i) acc += wr[i] * h[i];
    for (int i = 0; i < HD; ++i) acc += wr[HD + i] * ct[i];
    logits[tid] = acc;
  }
  __syncthreads();
  if (tid == 0) {
    float mx = logits[0];
    for (int o = 1; o < 26; ++o) mx = logits[o] > mx ? logits[o] : mx;
    float e[26], sm = 0.f;
    for (int o = 0; o < 26; ++o) { e[o] = __expf(logits[o] - mx); sm += e[o]; }
    float inv = 1.f / sm;
    for (int o = 0; o < 26; ++o) preds[s * 26 + o] = e[o] * inv;
  }
  __syncthreads();
  if (tid < HD) Hbuf[tid] = h[tid];   // carry h_new to next outer step
}

// ---------------------------------------------------------------------------
extern "C" void kernel_launch(void* const* d_in, const int* in_sizes, int n_in,
                              void* d_out, int out_size, void* d_ws, size_t ws_size,
                              hipStream_t stream) {
  const float* X      = (const float*)d_in[0];
  const float* w1     = (const float*)d_in[1];
  const float* b1     = (const float*)d_in[2];
  const float* w2     = (const float*)d_in[3];
  const float* b2     = (const float*)d_in[4];
  const float* ewih   = (const float*)d_in[5];
  const float* ewhh   = (const float*)d_in[6];
  const float* ebih   = (const float*)d_in[7];
  const float* ebhh   = (const float*)d_in[8];
  const float* dwih   = (const float*)d_in[9];
  const float* dwhh   = (const float*)d_in[10];
  const float* dbih   = (const float*)d_in[11];
  const float* dbhh   = (const float*)d_in[12];
  const float* fcw    = (const float*)d_in[13];
  const float* fcb    = (const float*)d_in[14];
  float* preds = (float*)d_out;

  // workspace layout (~3.1 MB)
  float* ws      = (float*)d_ws;
  float* P       = ws;                   // 768
  float* Q       = P + GD;               // 768
  float* Hbuf    = Q + GD;               // 256
  float* WhhT_e  = Hbuf + HD;            // 196608
  float* WhhT_d  = WhhT_e + GD * HD;     // 196608
  float* XP      = WhhT_d + GD * HD;     // 196608
  float* DecOut  = XP + GD * HD;         // 65536
  _Float16* Wih_h = (_Float16*)(DecOut + TT * HD);   // 196608 halves
  _Float16* Xc_h  = Wih_h + GD * HD;                 // 65536 halves

  front_kernel<<<1, GD, 0, stream>>>(w1, b1, w2, b2, ewih, ebih, P, Q);
  prep_kernel<<<(GD * HD + 255) / 256, 256, 0, stream>>>(ewhh, dwhh, dwih,
                                                         WhhT_e, WhhT_d, Wih_h);
  encoder_kernel<<<1, GD, 0, stream>>>(X, P, Q, WhhT_e, ebhh, Xc_h, Hbuf);

  for (int s = 0; s < TT; ++s) {
    dec_gemm_kernel<<<96, 256, 0, stream>>>(Xc_h, Wih_h, dbih, XP);
    dec_step_kernel<<<1, GD, 0, stream>>>(XP, WhhT_d, dbhh, DecOut, Xc_h, Hbuf,
                                          fcw, fcb, preds, s);
  }
}
